// GAT_Net_4286377361510
// MI455X (gfx1250) — matrix operand order repaired and checked by reading, not compiled, not run on hardware
//
#include <hip/hip_runtime.h>
#include <math.h>

// ---------------------------------------------------------------------------
// GAT 2-layer network for MI455X (gfx1250), wave32 + WMMA.
// Problem constants from the reference:
//   N=50000 nodes, F_in=256, layer1: H=4,C=64 (H*C=256), layer2: H=1,C=16
//   E0=800000 random edges + 50000 self loops = 850000 edges
// GEMMs run through v_wmma_f32_16x16x32_f16 (f16 in, f32 accumulate).
// Segment softmax / aggregation done with L2 float atomics (working set
// ~130MB fits the 192MB L2). Workspace use ~145MB with region reuse.
// ---------------------------------------------------------------------------

typedef __attribute__((ext_vector_type(16))) _Float16 v16h;
typedef __attribute__((ext_vector_type(8)))  float    v8f;

#define NODES     50000
#define FIN       256
#define HC1       256     // layer1 H*C
#define H1N       4
#define C1N       64
#define F2        16
#define E0EDGES   800000
#define ETOT      850000
#define NEG_SLOPE 0.2f

// ---------------------------------------------------------------- helpers ---
__device__ __forceinline__ void edge_sd(const int* __restrict__ ei, int e,
                                        int& s, int& d) {
  if (e < E0EDGES) { s = ei[e]; d = ei[E0EDGES + e]; }
  else             { s = e - E0EDGES; d = s; }          // self loops appended
}

// float atomic max via sign-split int/uint trick (init to -3.0e38f)
__device__ __forceinline__ void atomicMaxF(float* addr, float val) {
  if (val >= 0.0f) atomicMax((int*)addr, __float_as_int(val));
  else             atomicMin((unsigned int*)addr, __float_as_uint(val));
}

// ---------------------------------------------------------------- kernels ---
__global__ void k_f32_to_f16(const float* __restrict__ in,
                             _Float16* __restrict__ out, int n) {
  int i = blockIdx.x * blockDim.x + threadIdx.x;
  if (i < n) out[i] = (_Float16)in[i];
}

__global__ void k_fill(float* __restrict__ p, float v, int n) {
  int i = blockIdx.x * blockDim.x + threadIdx.x;
  if (i < n) p[i] = v;
}

// Pack row-major W[K x Ncols] (f32) into per-wave B fragments (f16):
// layout [ntile][kstep][lane][16 halves]; lane holds column n = ntile*16 +
// (lane&15), elements j = W[kstep*32 + (lane>>4)*16 + j, n].
__global__ void k_pack_b(const float* __restrict__ W, _Float16* __restrict__ pk,
                         int Ncols, int ksteps) {
  int tid = blockIdx.x * blockDim.x + threadIdx.x;
  int total = (Ncols / 16) * ksteps * 32;
  if (tid >= total) return;
  int lane  = tid & 31;
  int ks    = (tid >> 5) % ksteps;
  int ntile = tid / (32 * ksteps);
  int n  = ntile * 16 + (lane & 15);
  int kb = ks * 32 + (lane >> 4) * 16;
  _Float16* dst = pk + (size_t)tid * 16;
#pragma unroll
  for (int j = 0; j < 16; ++j)
    dst[j] = (_Float16)W[(size_t)(kb + j) * Ncols + n];
}

// Tiled f16 WMMA GEMM: C[M x N] = A[M x K] * B[K x N], f32 accumulate.
// One wave per 16x16 C tile; grid.x = M/16, (grid.y * waves) = N/16.
// M and K multiples of 16/32 -> EXEC is all-1s around every WMMA.
__global__ void __launch_bounds__(128)
k_wmma_gemm(const _Float16* __restrict__ A, const _Float16* __restrict__ Bp,
            float* __restrict__ C, int lda, int ldc, int ntiles, int ksteps) {
  int wave  = threadIdx.x >> 5;
  int ntile = blockIdx.y * (blockDim.x >> 5) + wave;
  if (ntile >= ntiles) return;
  int lane = threadIdx.x & 31;
  int lr = lane & 15, lh = lane >> 4;
  int mtile = blockIdx.x;

  const _Float16* arow  = A  + (size_t)(mtile * 16 + lr) * lda;
  const _Float16* bbase = Bp + (size_t)ntile * ksteps * 32 * 16;

  v8f acc = {};
  for (int ks = 0; ks < ksteps; ++ks) {
    int kbase = ks * 32 + lh * 16;
    union { v16h h; float4 f[2]; } a, b;
    // A 16x32 f16 fragment: lane row = lr, K contiguous [kbase,+16)
    a.f[0] = *(const float4*)(arow + kbase);
    a.f[1] = *(const float4*)(arow + kbase + 8);
    // B fragment: contiguous 32B per lane from the packed buffer
    const _Float16* bp = bbase + ((size_t)ks * 32 + lane) * 16;
    b.f[0] = ((const float4*)bp)[0];
    b.f[1] = ((const float4*)bp)[1];
    // prefetch next k-step of A (streaming dimension) -> global_prefetch_b8
    __builtin_prefetch(arow + kbase + 32, 0, 3);
    acc = __builtin_amdgcn_wmma_f32_16x16x32_f16(
        false, a.h, false, b.h, (short)0, acc, false, false);
  }
  // C layout: element r -> row (lh*8 + r), col lr
  float* crow = C + (size_t)(mtile * 16 + lh * 8) * ldc + ntile * 16 + lr;
#pragma unroll
  for (int r = 0; r < 8; ++r) crow[(size_t)r * ldc] = acc[r];
}

// a_src[n,h] = <h[n,h,:], att_src[h,:]>, a_dst likewise (one h1 read for both)
__global__ void k_attn(const float* __restrict__ h, const float* __restrict__ atts,
                       const float* __restrict__ attd, float* __restrict__ as,
                       float* __restrict__ ad, int Hn, int Cn) {
  int i = blockIdx.x * blockDim.x + threadIdx.x;
  if (i >= NODES * Hn) return;
  int n = i / Hn, hh = i % Hn;
  const float* hp = h + (size_t)n * Hn * Cn + (size_t)hh * Cn;
  const float* sp = atts + hh * Cn;
  const float* dp = attd + hh * Cn;
  float s1 = 0.0f, s2 = 0.0f;
  for (int c = 0; c < Cn; ++c) { float v = hp[c]; s1 += v * sp[c]; s2 += v * dp[c]; }
  as[i] = s1; ad[i] = s2;
}

__global__ void k_edge_max(const int* __restrict__ ei, const float* __restrict__ as,
                           const float* __restrict__ ad, float* __restrict__ m, int Hn) {
  int e = blockIdx.x * blockDim.x + threadIdx.x;
  if (e >= ETOT) return;
  int s, d; edge_sd(ei, e, s, d);
  for (int h = 0; h < Hn; ++h) {
    float v = as[s * Hn + h] + ad[d * Hn + h];
    v = (v >= 0.0f) ? v : NEG_SLOPE * v;
    atomicMaxF(&m[d * Hn + h], v);
  }
}

__global__ void k_edge_exp(const int* __restrict__ ei, const float* __restrict__ as,
                           const float* __restrict__ ad, const float* __restrict__ m,
                           float* __restrict__ t, float* __restrict__ dn, int Hn) {
  int e = blockIdx.x * blockDim.x + threadIdx.x;
  if (e >= ETOT) return;
  int s, d; edge_sd(ei, e, s, d);
  for (int h = 0; h < Hn; ++h) {
    float v = as[s * Hn + h] + ad[d * Hn + h];
    v = (v >= 0.0f) ? v : NEG_SLOPE * v;
    float tv = expf(v - m[d * Hn + h]);
    t[e * Hn + h] = tv;
    atomicAdd(&dn[d * Hn + h], tv);
  }
}

// layer1 aggregation: 64 threads/edge, each moves one float4 of the 256-vec
__global__ void k_scatter1(const int* __restrict__ ei, const float* __restrict__ h1,
                           const float* __restrict__ t, const float* __restrict__ dn,
                           float* __restrict__ out) {
  int gid = blockIdx.x * blockDim.x + threadIdx.x;
  int e = gid >> 6;
  if (e >= ETOT) return;
  int j = gid & 63;                 // float4 slot; head = j/16
  int s, d; edge_sd(ei, e, s, d);
  int h = j >> 4;
  float coef = t[e * 4 + h] / (dn[d * 4 + h] + 1e-16f);
  float4 v = *(const float4*)(h1 + (size_t)s * HC1 + j * 4);
  float* o = out + (size_t)d * HC1 + j * 4;
  atomicAdd(o + 0, v.x * coef);
  atomicAdd(o + 1, v.y * coef);
  atomicAdd(o + 2, v.z * coef);
  atomicAdd(o + 3, v.w * coef);
}

// oh = f16(relu(acc + b1)), the GEMM2 input (no f32 output)
__global__ void k_bias_relu_f16(const float* __restrict__ acc,
                                const float* __restrict__ b,
                                _Float16* __restrict__ oh, int n, int ld) {
  int i = blockIdx.x * blockDim.x + threadIdx.x;
  if (i >= n) return;
  float v = acc[i] + b[i % ld];
  oh[i] = (_Float16)fmaxf(v, 0.0f);
}

// layer2 aggregation: 16 threads/edge, one float each
__global__ void k_scatter2(const int* __restrict__ ei, const float* __restrict__ g2,
                           const float* __restrict__ t, const float* __restrict__ dn,
                           float* __restrict__ z2) {
  int gid = blockIdx.x * blockDim.x + threadIdx.x;
  int e = gid >> 4;
  if (e >= ETOT) return;
  int c = gid & 15;
  int s, d; edge_sd(ei, e, s, d);
  float coef = t[e] / (dn[d] + 1e-16f);
  atomicAdd(&z2[d * F2 + c], g2[(size_t)s * F2 + c] * coef);
}

__global__ void k_logsoftmax(const float* __restrict__ z2, const float* __restrict__ b2,
                             float* __restrict__ out) {
  int n = blockIdx.x * blockDim.x + threadIdx.x;
  if (n >= NODES) return;
  float z[F2];
  float mx = -3.0e38f;
#pragma unroll
  for (int c = 0; c < F2; ++c) { z[c] = z2[n * F2 + c] + b2[c]; mx = fmaxf(mx, z[c]); }
  float sum = 0.0f;
#pragma unroll
  for (int c = 0; c < F2; ++c) sum += expf(z[c] - mx);
  float l = logf(sum) + mx;
#pragma unroll
  for (int c = 0; c < F2; ++c) out[n * F2 + c] = z[c] - l;
}

// ----------------------------------------------------------------- launch ---
static inline int cdiv(long long a, int b) { return (int)((a + b - 1) / b); }

extern "C" void kernel_launch(void* const* d_in, const int* in_sizes, int n_in,
                              void* d_out, int out_size, void* d_ws, size_t ws_size,
                              hipStream_t stream) {
  const float* x    = (const float*)d_in[0];
  const int*   ei   = (const int*)d_in[1];   // [2, 800000], row0=src, row1=dst
  const float* W1   = (const float*)d_in[2];
  const float* at_s1 = (const float*)d_in[3];
  const float* at_d1 = (const float*)d_in[4];
  const float* b1   = (const float*)d_in[5];
  const float* W2   = (const float*)d_in[6];
  const float* at_s2 = (const float*)d_in[7];
  const float* at_d2 = (const float*)d_in[8];
  const float* b2   = (const float*)d_in[9];
  float* out = (float*)d_out;

  // workspace carve-up (256B aligned); total ~145MB with reuse
  char* ws = (char*)d_ws;
  size_t off = 0;
  auto take = [&](size_t bytes) -> char* {
    char* p = ws + off;
    off = (off + bytes + 255) & ~(size_t)255;
    return p;
  };
  _Float16* xh   = (_Float16*)take((size_t)NODES * FIN * 2);  // later reused as f16 relu(h1)
  _Float16* w1p  = (_Float16*)take(256 * 256 * 2);
  _Float16* w2p  = (_Float16*)take(256 * 16 * 2);
  float* h1   = (float*)take((size_t)NODES * HC1 * 4);        // region reused for layer-2 bufs
  float* as1  = (float*)take((size_t)NODES * H1N * 4);
  float* ad1  = (float*)take((size_t)NODES * H1N * 4);
  float* m1   = (float*)take((size_t)NODES * H1N * 4);
  float* dn1  = (float*)take((size_t)NODES * H1N * 4);
  float* t1   = (float*)take((size_t)ETOT * H1N * 4);
  float* out1 = (float*)take((size_t)NODES * HC1 * 4);

  // layer-2 buffers live inside the h1 region (h1 dead after scatter1)
  char* l2 = (char*)h1;
  auto take2 = [&](size_t bytes) -> char* {
    char* p = l2;
    l2 += (bytes + 255) & ~(size_t)255;
    return p;
  };
  float* g2  = (float*)take2((size_t)NODES * F2 * 4);
  float* as2 = (float*)take2((size_t)NODES * 4);
  float* ad2 = (float*)take2((size_t)NODES * 4);
  float* m2  = (float*)take2((size_t)NODES * 4);
  float* dn2 = (float*)take2((size_t)NODES * 4);
  float* t2  = (float*)take2((size_t)ETOT * 4);
  float* z2  = (float*)take2((size_t)NODES * F2 * 4);

  const int B = 256;

  // ---- layer 1 ----
  int nX = NODES * FIN;
  k_f32_to_f16<<<cdiv(nX, B), B, 0, stream>>>(x, xh, nX);
  k_pack_b<<<cdiv(16 * 8 * 32, B), B, 0, stream>>>(W1, w1p, 256, 8);
  k_pack_b<<<1, 256, 0, stream>>>(W2, w2p, 16, 8);

  // h1 = x @ W1  (3125 M-tiles x 16 N-tiles, one wave each)
  k_wmma_gemm<<<dim3(NODES / 16, 4), 128, 0, stream>>>(xh, w1p, h1, FIN, HC1, 16, 8);

  k_attn<<<cdiv(NODES * H1N, B), B, 0, stream>>>(h1, at_s1, at_d1, as1, ad1, H1N, C1N);

  k_fill<<<cdiv(NODES * H1N, B), B, 0, stream>>>(m1, -3.0e38f, NODES * H1N);
  k_fill<<<cdiv(NODES * H1N, B), B, 0, stream>>>(dn1, 0.0f, NODES * H1N);
  k_fill<<<cdiv(nX, B), B, 0, stream>>>(out1, 0.0f, nX);

  k_edge_max<<<cdiv(ETOT, B), B, 0, stream>>>(ei, as1, ad1, m1, H1N);
  k_edge_exp<<<cdiv(ETOT, B), B, 0, stream>>>(ei, as1, ad1, m1, t1, dn1, H1N);
  k_scatter1<<<cdiv((long long)ETOT * 64, B), B, 0, stream>>>(ei, h1, t1, dn1, out1);

  // relu(out1 + b1) -> f16 (reuses xh storage)
  k_bias_relu_f16<<<cdiv(nX, B), B, 0, stream>>>(out1, b1, xh, nX, HC1);

  // ---- layer 2 ----
  // g2 = relu(h1') @ W2  (1 N-tile: one wave per block)
  k_wmma_gemm<<<dim3(NODES / 16, 1), 32, 0, stream>>>(xh, w2p, g2, HC1, F2, 1, 8);

  k_attn<<<cdiv(NODES, B), B, 0, stream>>>(g2, at_s2, at_d2, as2, ad2, 1, F2);

  k_fill<<<cdiv(NODES, B), B, 0, stream>>>(m2, -3.0e38f, NODES);
  k_fill<<<cdiv(NODES, B), B, 0, stream>>>(dn2, 0.0f, NODES);
  k_fill<<<cdiv(NODES * F2, B), B, 0, stream>>>(z2, 0.0f, NODES * F2);

  k_edge_max<<<cdiv(ETOT, B), B, 0, stream>>>(ei, as2, ad2, m2, 1);
  k_edge_exp<<<cdiv(ETOT, B), B, 0, stream>>>(ei, as2, ad2, m2, t2, dn2, 1);
  k_scatter2<<<cdiv((long long)ETOT * 16, B), B, 0, stream>>>(ei, g2, t2, dn2, z2);

  k_logsoftmax<<<cdiv(NODES, B), B, 0, stream>>>(z2, b2, out);
}
